// MiniTransformerAttention_50534585205271
// MI455X (gfx1250) — matrix operand order repaired and checked by reading, not compiled, not run on hardware
//
#include <hip/hip_runtime.h>
#include <hip/hip_bf16.h>

typedef __attribute__((ext_vector_type(16))) __bf16 v16bf;
typedef __attribute__((ext_vector_type(8)))  __bf16 v8bf;
typedef __attribute__((ext_vector_type(8)))  float  v8f;

static constexpr int Bc = 2, Sc = 2048, Hc = 1024, Dc = 64, NH = 16, Mc = Bc * Sc;

// Optional CDNA5 async global->LDS path (ASYNCcnt-tracked), compile-safe probe.
#if defined(__has_builtin)
#if __has_builtin(__builtin_amdgcn_global_load_async_to_lds_b128) && \
    __has_builtin(__builtin_amdgcn_s_wait_asynccnt)
#define USE_ASYNC_LDS 1
#endif
#endif

#ifdef USE_ASYNC_LDS
// Builtin param 1 (from hipcc diagnostic) is 'int __attribute__((vector_size(16))) *'.
typedef int v4i __attribute__((__vector_size__(4 * sizeof(int))));
#define ASYNC_CP16(src_glb, dst_lds) \
  __builtin_amdgcn_global_load_async_to_lds_b128((v4i*)(src_glb), (v4i*)(dst_lds), 0, 0)
#endif

// ---------------------------------------------------------------------------
// WMMA helpers (CDNA5 wave32, v_wmma_f32_16x16x32_bf16)
// ---------------------------------------------------------------------------
__device__ __forceinline__ v8f wmma_bf16(v16bf a, v16bf b, v8f c) {
  return __builtin_amdgcn_wmma_f32_16x16x32_bf16(
      /*neg_a=*/false, a, /*neg_b=*/false, b,
      /*c_mod=*/(short)0, c, /*reuse_a=*/false, /*reuse_b=*/false);
}

__device__ __forceinline__ v16bf concat16(v8bf lo, v8bf hi) {
  union { v16bf v; v8bf h[2]; } u;
  u.h[0] = lo; u.h[1] = hi;
  return u.v;
}

// A-matrix 16x32 bf16 fragment. Caller passes pointer to row m (m = lane&15)
// at k-offset k0. Lanes 0-15 hold K={0..7,16..23}, lanes 16-31 K={8..15,24..31}.
__device__ __forceinline__ v16bf load_a_frag(const __bf16* row_k0, int lane) {
  const int khalf = lane >> 4;
  return concat16(*(const v8bf*)(row_k0 + 8 * khalf),
                  *(const v8bf*)(row_k0 + 16 + 8 * khalf));
}

// B-matrix 32x16 bf16 fragment. Lane n (= lane&15) holds column n,
// K = {0..7,16..23} (lanes 0-15) or {8..15,24..31} (lanes 16-31): two 16B reads of
// "row n" of a [N,K] row-major source.
__device__ __forceinline__ v16bf load_b_frag(const __bf16* row_k0, int lane) {
  const int khalf = lane >> 4;
  return concat16(*(const v8bf*)(row_k0 + 8 * khalf),
                  *(const v8bf*)(row_k0 + 16 + 8 * khalf));
}

// ---------------------------------------------------------------------------
// fp32 -> bf16 cast / additive-mask prep
// ---------------------------------------------------------------------------
__global__ __launch_bounds__(256) void cast_f32_bf16(const float* __restrict__ in,
                                                     __bf16* __restrict__ out, int n) {
  int i = blockIdx.x * 256 + threadIdx.x;
  if (i < n) out[i] = (__bf16)in[i];
}

__global__ __launch_bounds__(256) void mask_prep(const float* __restrict__ mask,
                                                 float* __restrict__ madd, int n) {
  int i = blockIdx.x * 256 + threadIdx.x;
  if (i < n) madd[i] = (1.0f - mask[i]) * -10000.0f;
}

// ---------------------------------------------------------------------------
// GEMM: out[m,n] = (sum_k A[m,k] * W[n,k] + bias[n]) * oscale
// A: [Mc,Hc] bf16 row-major, W: [Hc,Hc] bf16 row-major ([N,K]).
// HEADSPLIT: write bf16 into [B, head, S, d].  else: write fp32 [Mc,Hc].
// Block: 256 threads = 8 waves (2x4); wave tile 64x64 (4x4 WMMA accumulators)
// -> block tile 128x256.  Per K32 step: 8 fragment loads, 16 WMMAs.
// oscale = 1/sqrt(64) for the Q projection (folds the score scale in).
// ---------------------------------------------------------------------------
template <bool HEADSPLIT>
__global__ __launch_bounds__(256) void gemm_bf16(const __bf16* __restrict__ A,
                                                 const __bf16* __restrict__ W,
                                                 const float* __restrict__ bias,
                                                 __bf16* __restrict__ outb,
                                                 float* __restrict__ outf,
                                                 float oscale) {
  const int lane = threadIdx.x & 31;
  const int wid  = threadIdx.x >> 5;
  const int wm = wid & 1, wn = wid >> 1;          // 2 x 4 waves
  const int mblk = blockIdx.x * 128 + wm * 64;
  const int nblk = blockIdx.y * 256 + wn * 64;
  const int l15 = lane & 15;

  const __bf16* ar[4];
  const __bf16* br[4];
#pragma unroll
  for (int i = 0; i < 4; ++i) {
    ar[i] = A + (size_t)(mblk + 16 * i + l15) * Hc;
    br[i] = W + (size_t)(nblk + 16 * i + l15) * Hc;
  }

  v8f acc[4][4] = {};
  for (int k0 = 0; k0 < Hc; k0 += 32) {
    __builtin_prefetch(ar[0] + k0 + 256, 0, 0);   // global_prefetch_b8
    __builtin_prefetch(br[0] + k0 + 256, 0, 0);
    v16bf a[4], b[4];
#pragma unroll
    for (int i = 0; i < 4; ++i) a[i] = load_a_frag(ar[i] + k0, lane);
#pragma unroll
    for (int j = 0; j < 4; ++j) b[j] = load_b_frag(br[j] + k0, lane);
#pragma unroll
    for (int i = 0; i < 4; ++i)
#pragma unroll
      for (int j = 0; j < 4; ++j) acc[i][j] = wmma_bf16(a[i], b[j], acc[i][j]);
  }

  const int mbase = (lane >> 4) * 8;  // C/D layout: lane=col, vgpr r = row
#pragma unroll
  for (int ti = 0; ti < 4; ++ti) {
#pragma unroll
    for (int tj = 0; tj < 4; ++tj) {
      const int n = nblk + tj * 16 + l15;
      const float bv = bias[n];
#pragma unroll
      for (int r = 0; r < 8; ++r) {
        const int m = mblk + ti * 16 + mbase + r;
        const float v = (acc[ti][tj][r] + bv) * oscale;
        if (HEADSPLIT) {
          const int b = m >> 11, s = m & (Sc - 1);
          const int head = n >> 6, d = n & (Dc - 1);
          outb[((size_t)(b * NH + head) * Sc + s) * Dc + d] = (__bf16)v;
        } else {
          outf[(size_t)m * Hc + n] = v;
        }
      }
    }
  }
}

// ---------------------------------------------------------------------------
// Flash attention, 64-key tiles. One block = 64 query rows of one (b, head);
// 4 waves x 16 rows. Q fragments (pre-scaled by 1/sqrt(d)) live in VGPRs;
// K tile staged async to LDS; V tile staged TRANSPOSED so PV B-fragments are
// pairs of contiguous 16B LDS reads; ONE online-softmax update (one cross-lane max
// butterfly, one rescale) per 64 keys; row sums via P x ones WMMAs.
// Per 64-key tile: 8 score + 2 sum + 8 PV = 18 WMMAs.  LDS = 24KB.
// ---------------------------------------------------------------------------
__global__ __launch_bounds__(128) void flash_attn(const __bf16* __restrict__ Q,
                                                  const __bf16* __restrict__ Kb,
                                                  const __bf16* __restrict__ Vb,
                                                  const float* __restrict__ madd,
                                                  __bf16* __restrict__ ctx) {
  __shared__ __bf16 sK[64 * 64];     // [key][d]   row-major      (8KB)
  __shared__ __bf16 sVt[64 * 64];    // [d][key]   transposed     (8KB)
  __shared__ __bf16 sP[4 * 16 * 64]; // per-wave P tiles          (8KB)

  const int lane = threadIdx.x & 31;
  const int wid  = threadIdx.x >> 5;           // 0..3
  const int bh   = blockIdx.y;                 // b*16 + head
  const int b    = bh >> 4;
  const int head = bh & 15;
  const int qbase = blockIdx.x * 64 + wid * 16;
  const size_t headbase = (size_t)bh * Sc * Dc;

  const int l15   = lane & 15;
  const int mbase = (lane >> 4) * 8;

  // Q fragments for this wave's 16 rows: contraction over d=64 -> two K32 frags
  const __bf16* qrow = Q + headbase + (size_t)(qbase + l15) * Dc;
  v16bf qa0 = load_a_frag(qrow + 0, lane);
  v16bf qa1 = load_a_frag(qrow + 32, lane);

  v16bf ones;
#pragma unroll
  for (int i = 0; i < 16; ++i) ones[i] = (__bf16)1.0f;

  float rowM[8], rowL[8];
#pragma unroll
  for (int r = 0; r < 8; ++r) { rowM[r] = -1e30f; rowL[r] = 0.f; }
  v8f acc[4] = {};

  __bf16* pmine = sP + wid * (16 * 64);
  const int t = threadIdx.x;

  for (int kb = 0; kb < Sc; kb += 64) {
    const __bf16* gk = Kb + headbase + (size_t)kb * Dc;
    const __bf16* gv = Vb + headbase + (size_t)kb * Dc;

    // ---- K tile: 64x64 bf16 (8KB), ASYNCcnt-tracked b128 copies ----
#pragma unroll
    for (int c = 0; c < 4; ++c) {
#ifdef USE_ASYNC_LDS
      ASYNC_CP16(gk + (size_t)(t + c * 128) * 8, sK + (size_t)(t + c * 128) * 8);
#else
      ((uint4*)sK)[t + c * 128] = ((const uint4*)gk)[t + c * 128];
#endif
    }

    // ---- V tile: transpose 64x64 -> sVt[64 d][64 key] through VGPRs ----
#pragma unroll
    for (int c = 0; c < 4; ++c) {
      const int chunk = t + c * 128;           // 512 chunks of 8 bf16
      uint4 raw = ((const uint4*)gv)[chunk];
      const __bf16* e = (const __bf16*)&raw;
      const int key = chunk >> 3;              // 8 chunks per key row
      const int d0  = (chunk & 7) * 8;
#pragma unroll
      for (int j = 0; j < 8; ++j) sVt[(d0 + j) * 64 + key] = e[j];
    }

#ifdef USE_ASYNC_LDS
    __builtin_amdgcn_s_wait_asynccnt(0);
#endif
    __syncthreads();

    // ---- scores for four 16-key subtiles (8 WMMAs) ----
    v8f z = {};
    v8f sc[4];
    float mk[4];
#pragma unroll
    for (int s = 0; s < 4; ++s) {
      const __bf16* krow = sK + (s * 16 + l15) * Dc;
      sc[s] = wmma_bf16(qa1, load_b_frag(krow + 32, lane),
                        wmma_bf16(qa0, load_b_frag(krow + 0, lane), z));
      mk[s] = madd[(size_t)b * Sc + kb + s * 16 + l15];
    }

    // ---- single online-softmax update per 64-key tile ----
#pragma unroll
    for (int r = 0; r < 8; ++r) {
      float sv[4];
#pragma unroll
      for (int s = 0; s < 4; ++s) sv[s] = sc[s][r] + mk[s];
      float mx = fmaxf(fmaxf(sv[0], sv[1]), fmaxf(sv[2], sv[3]));
#pragma unroll
      for (int o = 8; o >= 1; o >>= 1) mx = fmaxf(mx, __shfl_xor(mx, o, 16));
      const float newM = fmaxf(rowM[r], mx);
      const float rescale = __expf(rowM[r] - newM);
      rowM[r] = newM;
      rowL[r] *= rescale;
#pragma unroll
      for (int d = 0; d < 4; ++d) acc[d][r] *= rescale;
#pragma unroll
      for (int s = 0; s < 4; ++s)
        pmine[(mbase + r) * 64 + s * 16 + l15] = (__bf16)__expf(sv[s] - newM);
    }

    // ---- P as two A-fragments; row sums via P x ones; PV (10 WMMAs) ----
    v16bf pa0 = load_a_frag(pmine + l15 * 64 + 0, lane);
    v16bf pa1 = load_a_frag(pmine + l15 * 64 + 32, lane);
    v8f ls = wmma_bf16(pa1, ones, wmma_bf16(pa0, ones, z));
#pragma unroll
    for (int r = 0; r < 8; ++r) rowL[r] += ls[r];
#pragma unroll
    for (int d = 0; d < 4; ++d) {
      const __bf16* vrow = sVt + (d * 16 + l15) * 64;
      acc[d] = wmma_bf16(pa1, load_b_frag(vrow + 32, lane),
                         wmma_bf16(pa0, load_b_frag(vrow + 0, lane), acc[d]));
    }
    __syncthreads();
  }

  // finalize: divide by row sums, write ctx bf16 [B,S, head*64 + d]
#pragma unroll
  for (int r = 0; r < 8; ++r) {
    const float inv = 1.0f / rowL[r];
    const int s = qbase + mbase + r;
    const size_t row = (size_t)(b * Sc + s);
#pragma unroll
    for (int d = 0; d < 4; ++d) {
      const int col = head * Dc + d * 16 + l15;
      ctx[row * Hc + col] = (__bf16)(acc[d][r] * inv);
    }
  }
}

// ---------------------------------------------------------------------------
// launch
// ---------------------------------------------------------------------------
extern "C" void kernel_launch(void* const* d_in, const int* in_sizes, int n_in,
                              void* d_out, int out_size, void* d_ws, size_t ws_size,
                              hipStream_t stream) {
  (void)in_sizes; (void)n_in; (void)out_size; (void)ws_size;
  const float* hidden = (const float*)d_in[0];
  const float* mask   = (const float*)d_in[1];
  const float* Wq = (const float*)d_in[2];
  const float* bq = (const float*)d_in[3];
  const float* Wk = (const float*)d_in[4];
  const float* bk = (const float*)d_in[5];
  const float* Wv = (const float*)d_in[6];
  const float* bv = (const float*)d_in[7];
  const float* Wo = (const float*)d_in[8];
  const float* bo = (const float*)d_in[9];
  float* out = (float*)d_out;

  const size_t MH = (size_t)Mc * Hc;   // 4096*1024
  const size_t HH = (size_t)Hc * Hc;   // 1024*1024
  __bf16* Xb  = (__bf16*)d_ws;
  __bf16* Wqb = Xb + MH;
  __bf16* Wkb = Wqb + HH;
  __bf16* Wvb = Wkb + HH;
  __bf16* Wob = Wvb + HH;
  __bf16* Qb  = Wob + HH;              // [B,h,S,d], pre-scaled by 1/8
  __bf16* Kbf = Qb + MH;
  __bf16* Vbf = Kbf + MH;
  __bf16* Ctx = Vbf + MH;              // [B,S,H]
  float*  Madd = (float*)(Ctx + MH);   // [B,S] additive mask

  // casts to bf16 + mask prep
  cast_f32_bf16<<<(int)((MH + 255) / 256), 256, 0, stream>>>(hidden, Xb, (int)MH);
  cast_f32_bf16<<<(int)((HH + 255) / 256), 256, 0, stream>>>(Wq, Wqb, (int)HH);
  cast_f32_bf16<<<(int)((HH + 255) / 256), 256, 0, stream>>>(Wk, Wkb, (int)HH);
  cast_f32_bf16<<<(int)((HH + 255) / 256), 256, 0, stream>>>(Wv, Wvb, (int)HH);
  cast_f32_bf16<<<(int)((HH + 255) / 256), 256, 0, stream>>>(Wo, Wob, (int)HH);
  mask_prep<<<(Bc * Sc) / 256, 256, 0, stream>>>(mask, Madd, Bc * Sc);

  // QKV projections -> [B,h,S,d] bf16 (Q folded with 1/sqrt(64))
  dim3 ggrid(Mc / 128, Hc / 256);
  gemm_bf16<true><<<ggrid, 256, 0, stream>>>(Xb, Wqb, bq, Qb, nullptr, 0.125f);
  gemm_bf16<true><<<ggrid, 256, 0, stream>>>(Xb, Wkb, bk, Kbf, nullptr, 1.0f);
  gemm_bf16<true><<<ggrid, 256, 0, stream>>>(Xb, Wvb, bv, Vbf, nullptr, 1.0f);

  // flash attention -> ctx bf16 [B,S,H]
  flash_attn<<<dim3(Sc / 64, Bc * NH), 128, 0, stream>>>(Qb, Kbf, Vbf, Madd, Ctx);

  // output projection -> fp32 out
  gemm_bf16<false><<<ggrid, 256, 0, stream>>>(Ctx, Wob, bo, nullptr, out, 1.0f);
}
